// SparseCrossAttention_20529943675444
// MI455X (gfx1250) — matrix operand order repaired and checked by reading, not compiled, not run on hardware
//
#include <hip/hip_runtime.h>
#include <cstdint>

// ---------------- problem constants ----------------
#define NTOK 16384
#define CDIM 512
#define BDIM 4
#define LDIM 512
#define CTXD 1024
#define HHEADS 8
#define DHEAD 64

typedef __bf16 bf16_t;
typedef bf16_t v16bf __attribute__((ext_vector_type(16)));
typedef bf16_t v2bf  __attribute__((ext_vector_type(2)));
typedef float  v8f   __attribute__((ext_vector_type(8)));
typedef float  v2f   __attribute__((ext_vector_type(2)));

union Frag {
    v16bf v;
    unsigned u[8];
    uint4 q[2];
};

// packed f32x2 -> bf16x2 (hardware v_cvt_pk_bf16_f32, RNE)
__device__ __forceinline__ unsigned pk2(float lo, float hi) {
    v2f f = {lo, hi};
    v2bf b = __builtin_convertvector(f, v2bf);
    return __builtin_bit_cast(unsigned, b);
}
__device__ __forceinline__ bf16_t f2bf(float x) {
    return (bf16_t)x;
}

__device__ __forceinline__ v8f wmma_bf16(const Frag& a, const Frag& b, v8f c) {
    return __builtin_amdgcn_wmma_f32_16x16x32_bf16(
        false, a.v, false, b.v, (short)0, c, false, false);
}

__device__ __forceinline__ v8f vzero8() {
    return (v8f){0.f, 0.f, 0.f, 0.f, 0.f, 0.f, 0.f, 0.f};
}

// A fragment (16x32 bf16) from f32 row-major source, K-chunk base kc.
// lanes 0-15: K = kc+{0..7, 16..23}; lanes 16-31: +8.
__device__ __forceinline__ Frag load_a_f32(const float* rowptr, int kc, bool hi) {
    int k0 = kc + (hi ? 8 : 0);
    float4 a0 = *(const float4*)(rowptr + k0);
    float4 a1 = *(const float4*)(rowptr + k0 + 4);
    float4 b0 = *(const float4*)(rowptr + k0 + 16);
    float4 b1 = *(const float4*)(rowptr + k0 + 20);
    Frag f;
    f.u[0] = pk2(a0.x, a0.y); f.u[1] = pk2(a0.z, a0.w);
    f.u[2] = pk2(a1.x, a1.y); f.u[3] = pk2(a1.z, a1.w);
    f.u[4] = pk2(b0.x, b0.y); f.u[5] = pk2(b0.z, b0.w);
    f.u[6] = pk2(b1.x, b1.y); f.u[7] = pk2(b1.z, b1.w);
    return f;
}

// A fragment from bf16 row-major source
__device__ __forceinline__ Frag load_a_bf16(const bf16_t* rowptr, int kc, bool hi) {
    const bf16_t* p = rowptr + kc + (hi ? 8 : 0);
    Frag f;
    f.q[0] = *(const uint4*)p;
    f.q[1] = *(const uint4*)(p + 16);
    return f;
}

// B fragment (32x16 bf16) from [N][K] bf16 storage (leading dim ldk).
// lanes 0-15: K = kc+{0..7, 16..23}; lanes 16-31: +8, N = lane&15.
__device__ __forceinline__ Frag load_b_bf16(const bf16_t* Bt, int ldk, int n, int kc, bool hi) {
    const bf16_t* p = Bt + (size_t)n * ldk + kc + (hi ? 8 : 0);
    Frag f;
    f.q[0] = *(const uint4*)p;
    f.q[1] = *(const uint4*)(p + 16);
    return f;
}

// ---------------- bucketing ----------------
__global__ void init_ws_kernel(int* counts, int* perm, int total) {
    int i = blockIdx.x * blockDim.x + threadIdx.x;
    if (i < 4) counts[i] = 0;
    if (i < total) perm[i] = -1;
}

__global__ void bucket_kernel(const int* bidx, int* counts, int* perm, int n) {
    int i = blockIdx.x * blockDim.x + threadIdx.x;
    if (i < n) {
        int b = bidx[i];
        int pos = atomicAdd(&counts[b], 1);
        perm[b * n + pos] = i;
    }
}

// ---------------- transpose + convert weights: f32 [K][Nc] -> bf16 [Nc][K] ----------------
__global__ __launch_bounds__(256) void transpose_cvt_kernel(const float* __restrict__ in,
                                                            bf16_t* __restrict__ out,
                                                            int K, int Nc) {
    __shared__ float tile[32][33];
    int n0 = blockIdx.x * 32, k0 = blockIdx.y * 32;
    int tx = threadIdx.x & 31, ty = threadIdx.x >> 5; // 32 x 8
    #pragma unroll
    for (int i = 0; i < 32; i += 8)
        tile[ty + i][tx] = in[(size_t)(k0 + ty + i) * Nc + n0 + tx];
    __syncthreads();
    #pragma unroll
    for (int i = 0; i < 32; i += 8)
        out[(size_t)(n0 + ty + i) * K + k0 + tx] = f2bf(tile[tx][ty + i]);
}

// ---------------- Q projection: q = x @ Wq + bq -> bf16 [N][C] ----------------
// wave tile: 16 (M) x 64 (N): A-fragment reused across 4 WMMAs
__global__ __launch_bounds__(256) void gemm_qproj_kernel(const float* __restrict__ A,
                                                         const bf16_t* __restrict__ WT,
                                                         const float* __restrict__ bias,
                                                         bf16_t* __restrict__ out) {
    int wave = threadIdx.x >> 5, lane = threadIdx.x & 31;
    int tile = blockIdx.x * 8 + wave;
    const int NG = CDIM / 64;  // 8
    int nt = tile % NG, mt = tile / NG;
    int m0 = mt * 16, n0 = nt * 64;
    bool hi = lane >= 16;
    int ml = lane & 15;
    const float* arow = A + (size_t)(m0 + ml) * CDIM;
    v8f acc[4] = {vzero8(), vzero8(), vzero8(), vzero8()};
    #pragma unroll 2
    for (int kc = 0; kc < CDIM; kc += 32) {
        Frag fa = load_a_f32(arow, kc, hi);
        #pragma unroll
        for (int t = 0; t < 4; t++) {
            Frag fb = load_b_bf16(WT, CDIM, n0 + t * 16 + ml, kc, hi);
            acc[t] = wmma_bf16(fa, fb, acc[t]);
        }
    }
    #pragma unroll
    for (int t = 0; t < 4; t++) {
        float bv = bias[n0 + t * 16 + ml];
        #pragma unroll
        for (int r = 0; r < 8; r++) {
            int row = m0 + r + (hi ? 8 : 0);
            out[(size_t)row * CDIM + n0 + t * 16 + ml] = f2bf(acc[t][r] + bv);
        }
    }
}

// ---------------- KV projection, scatter k[b][h][l][d], vT[b][h][d][l] ----------------
__global__ __launch_bounds__(256) void gemm_kvproj_kernel(const float* __restrict__ A,   // [2048][1024]
                                                          const bf16_t* __restrict__ WT, // [1024][1024]
                                                          const float* __restrict__ bias,
                                                          bf16_t* __restrict__ kws,
                                                          bf16_t* __restrict__ vT) {
    int wave = threadIdx.x >> 5, lane = threadIdx.x & 31;
    int tile = blockIdx.x * 8 + wave;
    const int NG = (2 * CDIM) / 64;  // 16
    int nt = tile % NG, mt = tile / NG;
    int m0 = mt * 16, n0 = nt * 64;
    bool hi = lane >= 16;
    int ml = lane & 15;
    const float* arow = A + (size_t)(m0 + ml) * CTXD;
    v8f acc[4] = {vzero8(), vzero8(), vzero8(), vzero8()};
    #pragma unroll 2
    for (int kc = 0; kc < CTXD; kc += 32) {
        Frag fa = load_a_f32(arow, kc, hi);
        #pragma unroll
        for (int t = 0; t < 4; t++) {
            Frag fb = load_b_bf16(WT, CTXD, n0 + t * 16 + ml, kc, hi);
            acc[t] = wmma_bf16(fa, fb, acc[t]);
        }
    }
    #pragma unroll
    for (int t = 0; t < 4; t++) {
        int nc = n0 + t * 16 + ml;
        float bv = bias[nc];
        #pragma unroll
        for (int r = 0; r < 8; r++) {
            int rr = m0 + r + (hi ? 8 : 0);
            int bb = rr >> 9, l = rr & 511;
            bf16_t ob = f2bf(acc[t][r] + bv);
            if (nc < CDIM) {
                int h = nc >> 6, d = nc & 63;
                kws[(((size_t)(bb * HHEADS + h) * LDIM) + l) * DHEAD + d] = ob;
            } else {
                int c2 = nc - CDIM;
                int h = c2 >> 6, d = c2 & 63;
                vT[(((size_t)(bb * HHEADS + h) * DHEAD) + d) * LDIM + l] = ob;
            }
        }
    }
}

// ---------------- fused attention ----------------
__global__ __launch_bounds__(128) void attention_kernel(const bf16_t* __restrict__ qws,   // [N][512]
                                                        const bf16_t* __restrict__ kws,   // [b][h][l][d]
                                                        const bf16_t* __restrict__ vT,    // [b][h][d][l]
                                                        const int* __restrict__ counts,
                                                        const int* __restrict__ perm,
                                                        bf16_t* __restrict__ aows) {      // [N][512]
    int b = blockIdx.y, h = blockIdx.z, tt = blockIdx.x;
    int cnt = counts[b];
    if (tt * 16 >= cnt) return;

    __shared__ int s_tok[16];
    __shared__ float s_red[4][16];
    __shared__ float s_final[16];
    __shared__ __align__(16) bf16_t sP[16][LDIM + 8];  // unnormalized probs

    int tid = threadIdx.x;
    int wave = tid >> 5, lane = tid & 31;
    int ml = lane & 15;
    bool hi = lane >= 16;

    if (tid < 16) s_tok[tid] = perm[b * NTOK + tt * 16 + tid];
    __syncthreads();

    // ---- Q fragments (gathered rows) ----
    int tok = s_tok[ml];
    Frag qa[2];
    #pragma unroll
    for (int c = 0; c < 2; c++) {
        if (tok >= 0) {
            qa[c] = load_a_bf16(qws + (size_t)tok * CDIM + h * DHEAD, c * 32, hi);
        } else {
            #pragma unroll
            for (int i = 0; i < 8; i++) qa[c].u[i] = 0u;
        }
    }

    // ---- logits: this wave covers l = wave*128 .. wave*128+127 ----
    const bf16_t* kb = kws + (size_t)(b * HHEADS + h) * LDIM * DHEAD;
    v8f acc[8];
    #pragma unroll
    for (int j = 0; j < 8; j++) acc[j] = vzero8();
    #pragma unroll
    for (int j = 0; j < 8; j++) {
        int ln = wave * 128 + j * 16 + ml;  // key row = B-fragment N index
        #pragma unroll
        for (int c = 0; c < 2; c++) {
            Frag fb = load_b_bf16(kb, DHEAD, ln, c * 32, hi);
            acc[j] = wmma_bf16(qa[c], fb, acc[j]);
        }
    }

    // ---- safe softmax (rows r + 8*hi) ----
    float rm[8];
    #pragma unroll
    for (int r = 0; r < 8; r++) {
        float m = acc[0][r];
        #pragma unroll
        for (int j = 1; j < 8; j++) m = fmaxf(m, acc[j][r]);
        #pragma unroll
        for (int off = 1; off < 16; off <<= 1) m = fmaxf(m, __shfl_xor(m, off, 32));
        rm[r] = m;
    }
    if (ml == 0) {
        int base = hi ? 8 : 0;
        #pragma unroll
        for (int r = 0; r < 8; r++) s_red[wave][base + r] = rm[r];
    }
    __syncthreads();
    if (tid < 16) {
        float m = s_red[0][tid];
        m = fmaxf(m, s_red[1][tid]);
        m = fmaxf(m, s_red[2][tid]);
        m = fmaxf(m, s_red[3][tid]);
        s_final[tid] = m;
    }
    __syncthreads();

    const float scale = 0.125f;  // D^-0.5
    float gmax[8], ssum[8];
    #pragma unroll
    for (int r = 0; r < 8; r++) { gmax[r] = s_final[r + (hi ? 8 : 0)]; ssum[r] = 0.f; }
    #pragma unroll
    for (int j = 0; j < 8; j++) {
        int col = wave * 128 + j * 16 + ml;
        #pragma unroll
        for (int r = 0; r < 8; r++) {
            int row = r + (hi ? 8 : 0);
            float e = __expf(scale * (acc[j][r] - gmax[r]));
            ssum[r] += e;
            sP[row][col] = f2bf(e);
        }
    }
    #pragma unroll
    for (int r = 0; r < 8; r++) {
        #pragma unroll
        for (int off = 1; off < 16; off <<= 1) ssum[r] += __shfl_xor(ssum[r], off, 32);
    }
    if (ml == 0) {
        int base = hi ? 8 : 0;
        #pragma unroll
        for (int r = 0; r < 8; r++) s_red[wave][base + r] = ssum[r];
    }
    __syncthreads();
    if (tid < 16)
        s_final[tid] = s_red[0][tid] + s_red[1][tid] + s_red[2][tid] + s_red[3][tid];
    __syncthreads();

    // ---- out = P @ V : this wave computes d-slice wave*16..+15 over K=512 ----
    const bf16_t* vb = vT + (size_t)(b * HHEADS + h) * DHEAD * LDIM;
    int dn = wave * 16 + ml;  // B-fragment N index within head
    v8f oacc = vzero8();
    #pragma unroll 4
    for (int kc = 0; kc < LDIM; kc += 32) {
        Frag fa;
        {
            const bf16_t* p = &sP[ml][kc + (hi ? 8 : 0)];
            fa.q[0] = *(const uint4*)p;
            fa.q[1] = *(const uint4*)(p + 16);
        }
        Frag fb = load_b_bf16(vb, LDIM, dn, kc, hi);
        oacc = wmma_bf16(fa, fb, oacc);
    }

    #pragma unroll
    for (int r = 0; r < 8; r++) {
        int row = r + (hi ? 8 : 0);
        int t = s_tok[row];
        if (t >= 0) {
            float inv = 1.0f / s_final[row];
            aows[(size_t)t * CDIM + h * DHEAD + dn] = f2bf(oacc[r] * inv);
        }
    }
}

// ---------------- output projection: out = aows(bf16) @ Wo + bo -> f32 ----------------
__global__ __launch_bounds__(256) void gemm_outproj_kernel(const bf16_t* __restrict__ A,
                                                           const bf16_t* __restrict__ WT,
                                                           const float* __restrict__ bias,
                                                           float* __restrict__ out) {
    int wave = threadIdx.x >> 5, lane = threadIdx.x & 31;
    int tile = blockIdx.x * 8 + wave;
    const int NG = CDIM / 64;  // 8
    int nt = tile % NG, mt = tile / NG;
    int m0 = mt * 16, n0 = nt * 64;
    bool hi = lane >= 16;
    int ml = lane & 15;
    const bf16_t* arow = A + (size_t)(m0 + ml) * CDIM;
    v8f acc[4] = {vzero8(), vzero8(), vzero8(), vzero8()};
    #pragma unroll 2
    for (int kc = 0; kc < CDIM; kc += 32) {
        Frag fa = load_a_bf16(arow, kc, hi);
        #pragma unroll
        for (int t = 0; t < 4; t++) {
            Frag fb = load_b_bf16(WT, CDIM, n0 + t * 16 + ml, kc, hi);
            acc[t] = wmma_bf16(fa, fb, acc[t]);
        }
    }
    #pragma unroll
    for (int t = 0; t < 4; t++) {
        float bv = bias[n0 + t * 16 + ml];
        #pragma unroll
        for (int r = 0; r < 8; r++) {
            int row = m0 + r + (hi ? 8 : 0);
            out[(size_t)row * CDIM + n0 + t * 16 + ml] = acc[t][r] + bv;
        }
    }
}

// ---------------- launch ----------------
extern "C" void kernel_launch(void* const* d_in, const int* in_sizes, int n_in,
                              void* d_out, int out_size, void* d_ws, size_t ws_size,
                              hipStream_t stream) {
    const float* x_feats = (const float*)d_in[0];   // [N][C]
    const float* context = (const float*)d_in[1];   // [B][L][CTX]
    const int*   bidx    = (const int*)d_in[2];     // [N]
    const float* Wq      = (const float*)d_in[3];   // [C][C]
    const float* bq      = (const float*)d_in[4];
    const float* Wkv     = (const float*)d_in[5];   // [CTX][2C]
    const float* bkv     = (const float*)d_in[6];
    const float* Wo      = (const float*)d_in[7];   // [C][C]
    const float* bo      = (const float*)d_in[8];
    float* out = (float*)d_out;

    // workspace layout (256-B aligned)
    char* ws = (char*)d_ws;
    size_t off = 0;
    auto take = [&](size_t bytes) { size_t r = off; off = (off + bytes + 255) & ~(size_t)255; return r; };
    int*    counts = (int*)(ws + take(4 * sizeof(int)));
    int*    perm   = (int*)(ws + take((size_t)BDIM * NTOK * sizeof(int)));
    bf16_t* WqT    = (bf16_t*)(ws + take((size_t)CDIM * CDIM * 2));
    bf16_t* WkvT   = (bf16_t*)(ws + take((size_t)(2 * CDIM) * CTXD * 2));
    bf16_t* WoT    = (bf16_t*)(ws + take((size_t)CDIM * CDIM * 2));
    bf16_t* qws    = (bf16_t*)(ws + take((size_t)NTOK * CDIM * 2));
    bf16_t* kws    = (bf16_t*)(ws + take((size_t)BDIM * HHEADS * LDIM * DHEAD * 2));
    bf16_t* vTws   = (bf16_t*)(ws + take((size_t)BDIM * HHEADS * DHEAD * LDIM * 2));
    bf16_t* aows   = (bf16_t*)(ws + take((size_t)NTOK * CDIM * 2));

    // 1. bucket tokens by batch
    int total = BDIM * NTOK;
    init_ws_kernel<<<(total + 255) / 256, 256, 0, stream>>>(counts, perm, total);
    bucket_kernel<<<(NTOK + 255) / 256, 256, 0, stream>>>(bidx, counts, perm, NTOK);

    // 2. transpose+convert weights to bf16 [N][K]
    transpose_cvt_kernel<<<dim3(CDIM / 32, CDIM / 32), 256, 0, stream>>>(Wq, WqT, CDIM, CDIM);
    transpose_cvt_kernel<<<dim3(2 * CDIM / 32, CTXD / 32), 256, 0, stream>>>(Wkv, WkvT, CTXD, 2 * CDIM);
    transpose_cvt_kernel<<<dim3(CDIM / 32, CDIM / 32), 256, 0, stream>>>(Wo, WoT, CDIM, CDIM);

    // 3. projections (16x64 wave tiles)
    gemm_qproj_kernel<<<(NTOK / 16) * (CDIM / 64) / 8, 256, 0, stream>>>(x_feats, WqT, bq, qws);
    gemm_kvproj_kernel<<<(BDIM * LDIM / 16) * (2 * CDIM / 64) / 8, 256, 0, stream>>>(context, WkvT, bkv, kws, vTws);

    // 4. fused attention
    attention_kernel<<<dim3(NTOK / 16, BDIM, HHEADS), 128, 0, stream>>>(qws, kws, vTws, counts, perm, aows);

    // 5. output projection
    gemm_outproj_kernel<<<(NTOK / 16) * (CDIM / 64) / 8, 256, 0, stream>>>(aows, WoT, bo, out);
}
